// MultiHeadAttention_4896262718019
// MI455X (gfx1250) — matrix operand order repaired and checked by reading, not compiled, not run on hardware
//
#include <hip/hip_runtime.h>
#include <hip/hip_bf16.h>
#include <stdint.h>

typedef __attribute__((ext_vector_type(16))) _Float16 v16h;
typedef __attribute__((ext_vector_type(8)))  float    v8f;
typedef __attribute__((ext_vector_type(4)))  uint32_t u32x4;
typedef __attribute__((ext_vector_type(8)))  uint32_t u32x8;

#define WMMA_F16(A, B, C) \
    __builtin_amdgcn_wmma_f32_16x16x32_f16(false, (A), false, (B), (short)0, (C), false, false)

// Problem constants
#define BB 2
#define SS 2048
#define DD 1024
#define HH 16
#define DK 64
#define MM (BB * SS)

// ---------------------------------------------------------------------------
// CDNA5 async global->LDS copy (ASYNCcnt-tracked, no VGPR round-trip).
// Low 32 bits of a generic LDS pointer are the LDS byte address.
// ---------------------------------------------------------------------------
__device__ __forceinline__ void async_ld_b128(const void* lds_ptr, const void* gptr) {
    unsigned l = (unsigned)(uintptr_t)lds_ptr;
    asm volatile("global_load_async_to_lds_b128 %0, %1, off"
                 :: "v"(l), "v"(gptr) : "memory");
}
__device__ __forceinline__ void wait_async0() {
    asm volatile("s_wait_asynccnt 0" ::: "memory");
}

// ---------------------------------------------------------------------------
// CDNA5 Tensor Data Mover: 2D tile (tileW x tileH elements, 2B elements)
// from global (row stride = strideElems) into contiguous LDS. D# per ISA
// CDNA5 ISA: group0 {count=1, lds_addr, global_addr,
// type=2}, group1 {data_size=2B, tensor dims, tile dims, dim0 stride}.
// Wave-uniform SGPR operands; tracked by TENSORcnt.
// ---------------------------------------------------------------------------
__device__ __forceinline__ void tdm_load_2d(const void* lds_ptr, const void* gptr,
                                            unsigned tileW, unsigned tileH,
                                            unsigned tensorW, unsigned tensorH,
                                            unsigned strideElems) {
    uint64_t ga = (uint64_t)(uintptr_t)gptr;
    unsigned la = (unsigned)(uintptr_t)lds_ptr;
    u32x4 g0;
    g0[0] = 1u;                                                   // count=1, user D#
    g0[1] = la;                                                   // lds_addr
    g0[2] = (uint32_t)ga;                                         // global_addr[31:0]
    g0[3] = (uint32_t)((ga >> 32) & 0x01FFFFFFu) | (2u << 30);    // addr[56:32] | type=2
    u32x8 g1;
    g1[0] = (1u << 16);                                           // data_size code 1 = 2B
    g1[1] = (tensorW & 0xFFFFu) << 16;                            // tensor_dim0 lo16
    g1[2] = ((tensorW >> 16) & 0xFFFFu) | ((tensorH & 0xFFFFu) << 16);
    g1[3] = ((tensorH >> 16) & 0xFFFFu) | ((tileW & 0xFFFFu) << 16);
    g1[4] = tileH & 0xFFFFu;                                      // tile_dim1 (tile_dim2=0)
    g1[5] = strideElems;                                          // tensor_dim0_stride lo32
    g1[6] = 0u;
    g1[7] = 0u;
    asm volatile("tensor_load_to_lds %0, %1" :: "s"(g0), "s"(g1) : "memory");
}

// ---------------------------------------------------------------------------
// fp32 -> fp16 convert
// ---------------------------------------------------------------------------
__global__ __launch_bounds__(256) void f32_to_f16_kernel(const float* __restrict__ src,
                                                         _Float16* __restrict__ dst, int n) {
    int i = blockIdx.x * 256 + threadIdx.x;
    if (i < n) dst[i] = (_Float16)src[i];
}

// ---------------------------------------------------------------------------
// Tiled WMMA GEMM:  C(M,N) = A(M,K) * W(N,K)^T + bias
//   mode 0: f32 row-major (M,N)            (output projection)
//   mode 1: f16 (B,H,S,DK)                 (V projection)
//   mode 2: RoPE + 1/sqrt(DK) f16 BHSD     (Q projection)
//   mode 3: RoPE f16 BHSD                  (K projection)
// 128 threads = 4 waves; block tile 128x64, K-step 32; wave tile 32x64
// (8 WMMAs per k-step); TDM double-buffered LDS staging issued by wave 0.
// ---------------------------------------------------------------------------
__global__ __launch_bounds__(128)
void gemm_wmma_kernel(const _Float16* __restrict__ A, const _Float16* __restrict__ W,
                      const float* __restrict__ bias, void* __restrict__ dstv,
                      int M, int N, int K, int mode, const float* __restrict__ rope) {
    __shared__ __attribute__((aligned(16))) _Float16 As[2][128][32]; // [buf][m][k]
    __shared__ __attribute__((aligned(16))) _Float16 Bs[2][64][32];  // [buf][n][k]

    const int tid  = threadIdx.x;
    const int wave = tid >> 5;
    const int lane = tid & 31;
    const int lh   = lane & 15;
    const int hi   = lane >> 4;
    const int m0   = blockIdx.y * 128;
    const int n0   = blockIdx.x * 64;

    // TDM staging: one descriptor for the 128x32 A tile, one for the 64x32 B tile.
    auto stage_tdm = [&](int buf, int k0) {
        tdm_load_2d(&As[buf][0][0], A + (size_t)m0 * K + k0, 32, 128, K, M, K);
        tdm_load_2d(&Bs[buf][0][0], W + (size_t)n0 * K + k0, 32, 64,  K, N, K);
    };

    v8f acc[2][4] = {{v8f{}, v8f{}, v8f{}, v8f{}}, {v8f{}, v8f{}, v8f{}, v8f{}}};
    const int steps = K >> 5;
    if (wave == 0) stage_tdm(0, 0);

    for (int t = 0; t < steps; ++t) {
        if (wave == 0) __builtin_amdgcn_s_wait_tensorcnt(0);
        __syncthreads();      // tile ready; prior reads of other buf complete
        const int buf = t & 1;
        if (t + 1 < steps && wave == 0) stage_tdm(buf ^ 1, (t + 1) * 32);

        v16h af[2];
        {
            const int kb = hi ? 8 : 0;
#pragma unroll
            for (int ms = 0; ms < 2; ++ms) {
                const int mrow = 32 * wave + ms * 16 + lh;
#pragma unroll
                for (int e = 0; e < 16; ++e)
                    af[ms][e] = As[buf][mrow][kb + (e < 8 ? e : e + 8)];
            }
        }
#pragma unroll
        for (int nsub = 0; nsub < 4; ++nsub) {
            v16h bf;
            const int ncol = nsub * 16 + lh;
            const int kb2  = hi ? 8 : 0;
#pragma unroll
            for (int e = 0; e < 16; ++e) bf[e] = Bs[buf][ncol][kb2 + (e < 8 ? e : e + 8)];
#pragma unroll
            for (int ms = 0; ms < 2; ++ms)
                acc[ms][nsub] = WMMA_F16(af[ms], bf, acc[ms][nsub]);
        }
    }

    // Epilogue. C/D layout: VGPR r -> row m = r + hi*8, col n = lh per 16-wide subtile.
    if (mode == 0) {
        float* dst = (float*)dstv;
#pragma unroll
        for (int nsub = 0; nsub < 4; ++nsub) {
            int   n  = n0 + nsub * 16 + lh;
            float bv = bias[n];
#pragma unroll
            for (int ms = 0; ms < 2; ++ms)
#pragma unroll
                for (int r = 0; r < 8; ++r) {
                    int m = m0 + 32 * wave + ms * 16 + r + hi * 8;
                    dst[(size_t)m * N + n] = acc[ms][nsub][r] + bv;
                }
        }
    } else {
        _Float16* dst    = (_Float16*)dstv;
        const float qscl = (mode == 2) ? 0.125f : 1.0f;  // 1/sqrt(64)
#pragma unroll
        for (int nsub = 0; nsub < 4; ++nsub) {
            int   n  = n0 + nsub * 16 + lh;
            float bv = bias[n];
            int   h  = n >> 6;
            int   dk = n & 63;
#pragma unroll
            for (int ms = 0; ms < 2; ++ms)
#pragma unroll
                for (int r = 0; r < 8; ++r) {
                    int   m   = m0 + 32 * wave + ms * 16 + r + hi * 8;
                    int   b   = m >> 11;
                    int   s   = m & 2047;
                    float val = acc[ms][nsub][r] + bv;
                    if (mode >= 2) {
                        float part = __shfl_xor(val, 1, 32);
                        float fr   = rope[s * (DK / 2) + (dk >> 1)];
                        float c    = __cosf(fr);
                        float sn   = __sinf(fr);
                        val = val * c + ((n & 1) ? part * sn : -part * sn);
                        val *= qscl;
                    }
                    dst[(((size_t)b * HH + h) * SS + s) * DK + dk] = (_Float16)val;
                }
        }
    }
}

// ---------------------------------------------------------------------------
// Flash attention: one block = (b, h, 64-query tile). 4 waves x 16 q-rows.
// Async double-buffered 32-key K/V tiles; online softmax in f32; analytic
// causal mask; ALiBi bias streamed from the provided (H,S,S) tensor.
// ---------------------------------------------------------------------------
__global__ __launch_bounds__(128)
void attn_wmma_kernel(const _Float16* __restrict__ Q, const _Float16* __restrict__ Kt,
                      const _Float16* __restrict__ V, const float* __restrict__ bias,
                      _Float16* __restrict__ ctx) {
    __shared__ __attribute__((aligned(16))) _Float16 Qs[64][64];       // [q][dk]
    __shared__ __attribute__((aligned(16))) _Float16 Ks[2][32][64];    // [buf][key][dk]
    __shared__ __attribute__((aligned(16))) _Float16 Vs[2][32][64];    // [buf][key][dk]
    __shared__ __attribute__((aligned(16))) _Float16 Ps[4][16][32];    // per-wave P staging

    const int tid  = threadIdx.x;
    const int wave = tid >> 5;
    const int lane = tid & 31;
    const int lh   = lane & 15;
    const int hi   = lane >> 4;
    const int q0   = blockIdx.x * 64;
    const int h    = blockIdx.y;
    const int b    = blockIdx.z;

    const size_t headBase = (((size_t)b * HH + h) * SS) * DK;
    const int    nchunks  = (q0 + 64) >> 5;   // causal: keys < q0+64

    auto stageKV = [&](int buf, int kb0) {
        const size_t baseK = headBase + (size_t)kb0 * DK;
#pragma unroll
        for (int i = 0; i < 2; ++i) {
            int idx = tid + i * 128, r = idx >> 3, sg = idx & 7;
            async_ld_b128(&Ks[buf][r][sg * 8], Kt + baseK + (size_t)r * DK + sg * 8);
        }
#pragma unroll
        for (int i = 0; i < 2; ++i) {
            int idx = tid + i * 128, r = idx >> 3, sg = idx & 7;
            async_ld_b128(&Vs[buf][r][sg * 8], V + baseK + (size_t)r * DK + sg * 8);
        }
    };

    // Q tile (64x64) async + first K/V tile.
#pragma unroll
    for (int i = 0; i < 4; ++i) {
        int idx = tid + i * 128, r = idx >> 3, sg = idx & 7;
        async_ld_b128(&Qs[r][sg * 8], Q + headBase + (size_t)(q0 + r) * DK + sg * 8);
    }
    stageKV(0, 0);
    wait_async0();
    __syncthreads();

    // Per-wave Q fragments for the two 32-wide K-dim chunks of DK=64.
    v16h qf[2];
    {
        const int mrow = 16 * wave + lh;
        const int kb   = hi ? 8 : 0;
#pragma unroll
        for (int dc = 0; dc < 2; ++dc)
#pragma unroll
            for (int e = 0; e < 16; ++e)
                qf[dc][e] = Qs[mrow][dc * 32 + kb + (e < 8 ? e : e + 8)];
    }

    v8f   o[4] = {v8f{}, v8f{}, v8f{}, v8f{}};
    float rm[8], rl[8];
#pragma unroll
    for (int r = 0; r < 8; ++r) { rm[r] = -__builtin_inff(); rl[r] = 0.0f; }

    for (int kc = 0; kc < nchunks; ++kc) {
        if (kc > 0) {
            wait_async0();
            __syncthreads();
        }
        const int buf = kc & 1;
        const int kb0 = kc * 32;
        if (kc + 1 < nchunks) stageKV(buf ^ 1, kb0 + 32);

        // Scores: two 16-key subtiles, each over DK via two K=32 WMMAs.
        v8f sc[2] = {v8f{}, v8f{}};
#pragma unroll
        for (int ks = 0; ks < 2; ++ks) {
            const int key = ks * 16 + lh;
            const int kb2 = hi ? 8 : 0;
#pragma unroll
            for (int dc = 0; dc < 2; ++dc) {
                v16h bf;   // B 32x16: element e -> d-row = kb2+(e<8?e:e+8), col = key
#pragma unroll
                for (int e = 0; e < 16; ++e) bf[e] = Ks[buf][key][dc * 32 + kb2 + (e < 8 ? e : e + 8)];
                sc[ks] = WMMA_F16(qf[dc], bf, sc[ks]);
            }
        }

        // ALiBi + causal mask + online softmax; write P to per-wave LDS.
#pragma unroll
        for (int r = 0; r < 8; ++r) {
            const int    q    = q0 + 16 * wave + r + hi * 8;
            const size_t brow = ((size_t)h * SS + q) * SS + kb0;
            float s0 = sc[0][r] + bias[brow + lh];
            float s1 = sc[1][r] + bias[brow + 16 + lh];
            if (kb0 + 32 < SS) __builtin_prefetch(bias + brow + 32, 0, 0);
            if (kb0 + lh > q)      s0 = -__builtin_inff();
            if (kb0 + 16 + lh > q) s1 = -__builtin_inff();

            float mx = fmaxf(s0, s1);
#pragma unroll
            for (int off = 1; off < 16; off <<= 1)
                mx = fmaxf(mx, __shfl_xor(mx, off, 32));

            float mnew  = fmaxf(rm[r], mx);
            float scale = __expf(rm[r] - mnew);
            float e0    = __expf(s0 - mnew);
            float e1    = __expf(s1 - mnew);
            float rsum  = e0 + e1;
#pragma unroll
            for (int off = 1; off < 16; off <<= 1)
                rsum += __shfl_xor(rsum, off, 32);

            rl[r] = rl[r] * scale + rsum;
            rm[r] = mnew;
#pragma unroll
            for (int nsub = 0; nsub < 4; ++nsub)
                o[nsub][r] = o[nsub][r] * scale;

            const int prow = r + hi * 8;
            Ps[wave][prow][lh]      = (_Float16)e0;
            Ps[wave][prow][16 + lh] = (_Float16)e1;
        }

        // DS ops from the same wave are in-order; wait for the P stores.
        asm volatile("s_wait_dscnt 0" ::: "memory");

        // P (16x32) x V (32x64).
        v16h pf;
        {
            const int kb = hi ? 8 : 0;
#pragma unroll
            for (int e = 0; e < 16; ++e)
                pf[e] = Ps[wave][lh][kb + (e < 8 ? e : e + 8)];
        }
#pragma unroll
        for (int nsub = 0; nsub < 4; ++nsub) {
            v16h vf;   // B 32x16: element e -> key = (hi?8:0)+(e<8?e:e+8), col dk = nsub*16+lh
            const int dk  = nsub * 16 + lh;
            const int kb2 = hi ? 8 : 0;
#pragma unroll
            for (int e = 0; e < 16; ++e) vf[e] = Vs[buf][kb2 + (e < 8 ? e : e + 8)][dk];
            o[nsub] = WMMA_F16(pf, vf, o[nsub]);
        }
    }

    // Normalize and store ctx in (B,S,H*DK) f16 layout for the output GEMM.
#pragma unroll
    for (int nsub = 0; nsub < 4; ++nsub) {
#pragma unroll
        for (int r = 0; r < 8; ++r) {
            int   q   = q0 + 16 * wave + r + hi * 8;
            int   dk  = nsub * 16 + lh;
            float val = o[nsub][r] / rl[r];
            ctx[((size_t)b * SS + q) * DD + h * DK + dk] = (_Float16)val;
        }
    }
}

// ---------------------------------------------------------------------------
// Host-side orchestration
// ---------------------------------------------------------------------------
extern "C" void kernel_launch(void* const* d_in, const int* in_sizes, int n_in,
                              void* d_out, int out_size, void* d_ws, size_t ws_size,
                              hipStream_t stream) {
    const float* x    = (const float*)d_in[0];
    // d_in[1] = mask (causal tril) -> handled analytically
    const float* bias = (const float*)d_in[2];   // (H,S,S)
    const float* rope = (const float*)d_in[3];   // (S,DK/2)
    const float* Wq   = (const float*)d_in[4];
    const float* bq   = (const float*)d_in[5];
    const float* Wk   = (const float*)d_in[6];
    const float* bk   = (const float*)d_in[7];
    const float* Wv   = (const float*)d_in[8];
    const float* bv   = (const float*)d_in[9];
    const float* Wo   = (const float*)d_in[10];
    const float* bo   = (const float*)d_in[11];

    char* ws = (char*)d_ws;
    const size_t MB = 1024 * 1024;
    _Float16* Xh   = (_Float16*)(ws + 0 * MB);
    _Float16* Wqh  = (_Float16*)(ws + 8 * MB);
    _Float16* Wkh  = (_Float16*)(ws + 10 * MB);
    _Float16* Wvh  = (_Float16*)(ws + 12 * MB);
    _Float16* Woh  = (_Float16*)(ws + 14 * MB);
    _Float16* Qh   = (_Float16*)(ws + 16 * MB);
    _Float16* Kh   = (_Float16*)(ws + 24 * MB);
    _Float16* Vh   = (_Float16*)(ws + 32 * MB);
    _Float16* Ctxh = (_Float16*)(ws + 40 * MB);

    const int nx = MM * DD;
    const int nw = DD * DD;
    f32_to_f16_kernel<<<(nx + 255) / 256, 256, 0, stream>>>(x,  Xh,  nx);
    f32_to_f16_kernel<<<(nw + 255) / 256, 256, 0, stream>>>(Wq, Wqh, nw);
    f32_to_f16_kernel<<<(nw + 255) / 256, 256, 0, stream>>>(Wk, Wkh, nw);
    f32_to_f16_kernel<<<(nw + 255) / 256, 256, 0, stream>>>(Wv, Wvh, nw);
    f32_to_f16_kernel<<<(nw + 255) / 256, 256, 0, stream>>>(Wo, Woh, nw);

    dim3 ggrid(DD / 64, MM / 128);     // (16, 32)
    gemm_wmma_kernel<<<ggrid, 128, 0, stream>>>(Xh, Wqh, bq, (void*)Qh, MM, DD, DD, 2, rope);
    gemm_wmma_kernel<<<ggrid, 128, 0, stream>>>(Xh, Wkh, bk, (void*)Kh, MM, DD, DD, 3, rope);
    gemm_wmma_kernel<<<ggrid, 128, 0, stream>>>(Xh, Wvh, bv, (void*)Vh, MM, DD, DD, 1, nullptr);

    dim3 agrid(SS / 64, HH, BB);       // (32, 16, 2)
    attn_wmma_kernel<<<agrid, 128, 0, stream>>>(Qh, Kh, Vh, bias, Ctxh);

    gemm_wmma_kernel<<<ggrid, 128, 0, stream>>>(Ctxh, Woh, bo, d_out, MM, DD, DD, 0, nullptr);
}
